// NRField_58007828300000
// MI455X (gfx1250) — matrix operand order repaired and checked by reading, not compiled, not run on hardware
//
#include <hip/hip_runtime.h>
#include <hip/hip_bf16.h>
#include <stdint.h>

// ---------------------------------------------------------------------------
// NRField fused forward for gfx1250 (MI455X), wave32 + v_wmma_f32_16x16x32_f16
//
//  - hash-grid encode (instant-ngp) fused with 5-layer MLP
//  - "transposed" GEMM chaining: out^T = W * in^T, so D fragments of one
//    layer become B fragments of the next via cvt_pk + one ds_swizzle SWAPX16.
//    Entire MLP is register resident: no LDS round-trips, no barriers.
//  - weights pre-swizzled to f16 WMMA A-fragment layout (contiguous 32B/lane)
//  - M-tile pairs with two accumulators + 1-step weight pipelining; epilogue
//    deferred one pair so pack/ReLU/swizzle co-executes under the next pair's
//    WMMAs; swizzles batched before their consumers to overlap DS latency
//  - ReLU applied packed (v_pk_max_num_f16) after the f16 pack (numerically
//    identical under RNE)
// ---------------------------------------------------------------------------

typedef __attribute__((ext_vector_type(16))) _Float16 v16h;
typedef __attribute__((ext_vector_type(2)))  _Float16 v2h;
typedef __attribute__((ext_vector_type(8)))  float    v8f;
typedef __attribute__((ext_vector_type(8)))  unsigned v8u;

#define WMMA_F16(A, B, C) \
  __builtin_amdgcn_wmma_f32_16x16x32_f16(false, (A), false, (B), (short)0, (C), false, false)

static constexpr int      TSIZE = 1 << 22;            // hashmap entries / level
static constexpr unsigned TMASK = (unsigned)(TSIZE - 1);
static constexpr unsigned P1 = 2654435761u;
static constexpr unsigned P2 = 805459861u;

// half-pack two f32 -> one dword (two f16)
__device__ __forceinline__ unsigned pkh(float a, float b) {
  v2h v; v[0] = (_Float16)a; v[1] = (_Float16)b;
  return __builtin_bit_cast(unsigned, v);
}

// packed f16 ReLU on a half2 dword -> v_pk_max_num_f16
__device__ __forceinline__ unsigned pk_relu(unsigned u) {
  v2h x = __builtin_bit_cast(v2h, u);
  v2h z = {(_Float16)0.0f, (_Float16)0.0f};
  return __builtin_bit_cast(unsigned, __builtin_elementwise_max(x, z));
}

// half-wave exchange lane <-> lane^16: ds_swizzle_b32, group-of-32 SWAPX16
// (offset = xor_mask 0x10 << 10 | and_mask 0x1f = 0x401F)
__device__ __forceinline__ unsigned swap16(unsigned u) {
  return (unsigned)__builtin_amdgcn_ds_swizzle((int)u, 0x401f);
}

// load one 32B weight fragment chunk for this lane
__device__ __forceinline__ v16h ldw(const _Float16* __restrict__ Wh, int frag, int lane) {
  return *reinterpret_cast<const v16h*>(Wh + (frag * 32 + lane) * 16);
}

// pack + ReLU + half-wave exchange: D tiles (pair) -> next-layer B frag.
// Phases separated so all 4 ds_swizzles issue before any consumer waits.
__device__ __forceinline__ v8u epilogue_frag(const v8f& a0, const v8f& a1, bool lo) {
  unsigned s0[4], s1[4];
#pragma unroll
  for (int j = 0; j < 4; ++j) {
    s0[j] = pk_relu(pkh(a0[2 * j], a0[2 * j + 1]));
    s1[j] = pk_relu(pkh(a1[2 * j], a1[2 * j + 1]));
  }
  unsigned send[4], recv[4];
#pragma unroll
  for (int d = 0; d < 4; ++d) send[d] = lo ? s1[d] : s0[d];
#pragma unroll
  for (int d = 0; d < 4; ++d) recv[d] = swap16(send[d]);
  v8u f;
#pragma unroll
  for (int d = 0; d < 4; ++d) {
    f[d]     = lo ? s0[d] : recv[d];   // halves 0..7
    f[d + 4] = lo ? recv[d] : s1[d];   // halves 8..15
  }
  return f;
}

// ---------------------------------------------------------------------------
// Weight swizzle: f32 [M_real, K_real] row-major -> f16 WMMA-A fragments.
// Fragment (mt, kt): 32 lanes x 16 halves contiguous. A-layout used here (contiguous k, same order as the B frags):
//   lane&15 = row M; lane>=16 selects K+16;
//   halves 0..15 -> K = kt*32 + hi*16 + j
// ---------------------------------------------------------------------------
__global__ void swz_weights(const float* __restrict__ W, _Float16* __restrict__ dst,
                            int Mr, int Kr, int ktiles, int total) {
  int t = blockIdx.x * 256 + threadIdx.x;
  if (t >= total) return;
  int j    = t & 15;
  int lane = (t >> 4) & 31;
  int tile = t >> 9;
  int kt   = tile % ktiles;
  int mt   = tile / ktiles;
  int hi   = lane >> 4;
  int feat = mt * 16 + (lane & 15);
  int k    = kt * 32 + hi * 16 + j;
  float v  = (feat < Mr && k < Kr) ? W[feat * Kr + k] : 0.0f;
  dst[t] = (_Float16)v;
}

// ---------------------------------------------------------------------------
// One hidden layer (transposed): out^T tiles D[mt] (16 feats x 16 rays),
// K = KT*32 input feats held in Bin frags. Emits next-layer B frags in Bout.
// Tile pairs interleaved; weight loads 1 step ahead; epilogue deferred 1 pair.
// ---------------------------------------------------------------------------
template <int KT>
__device__ __forceinline__ void mlp_layer(const _Float16* __restrict__ Wh,
                                          const float* __restrict__ bias,
                                          const v8u (&Bin)[8], v8u (&Bout)[8],
                                          bool lo, int lane) {
  v16h wa = ldw(Wh, 0 * KT, lane);     // (mt=0, kt=0)
  v16h wb = ldw(Wh, 1 * KT, lane);     // (mt=1, kt=0)
  v8f pa, pb;                          // previous pair's accumulators
#pragma unroll
  for (int mtp = 0; mtp < 8; ++mtp) {
    const int mt0 = 2 * mtp, mt1 = mt0 + 1;
    // bias preload into C: feat = mt*16 + (hi?8:0) + j, broadcast over rays
    const float4* bp0 = reinterpret_cast<const float4*>(bias + mt0 * 16 + (lo ? 0 : 8));
    const float4* bp1 = reinterpret_cast<const float4*>(bias + mt1 * 16 + (lo ? 0 : 8));
    float4 a0 = bp0[0], a1 = bp0[1], c0 = bp1[0], c1 = bp1[1];
    v8f acc0 = {a0.x, a0.y, a0.z, a0.w, a1.x, a1.y, a1.z, a1.w};
    v8f acc1 = {c0.x, c0.y, c0.z, c0.w, c1.x, c1.y, c1.z, c1.w};
#pragma unroll
    for (int kt = 0; kt < KT; ++kt) {
      // issue next step's weight loads before consuming current fragments
      v16h na = wa, nb = wb;
      if (kt + 1 < KT) {
        na = ldw(Wh, mt0 * KT + kt + 1, lane);
        nb = ldw(Wh, mt1 * KT + kt + 1, lane);
      } else if (mtp < 7) {
        na = ldw(Wh, (mt0 + 2) * KT, lane);
        nb = ldw(Wh, (mt1 + 2) * KT, lane);
      }
      const v16h bf = __builtin_bit_cast(v16h, Bin[kt]);
      acc0 = WMMA_F16(wa, bf, acc0);
      acc1 = WMMA_F16(wb, bf, acc1);
      wa = na; wb = nb;
    }
    // previous pair's epilogue runs in the shadow of this pair's WMMAs
    if (mtp > 0) Bout[mtp - 1] = epilogue_frag(pa, pb, lo);
    pa = acc0; pb = acc1;
  }
  Bout[7] = epilogue_frag(pa, pb, lo);
}

// ---------------------------------------------------------------------------
// Fused kernel: 256 threads = 8 waves; each wave owns 16 rays.
// ---------------------------------------------------------------------------
__launch_bounds__(256, 1)
__global__ void nrfield_fused(const float* __restrict__ x,  const float* __restrict__ wi,
                              const float* __restrict__ nn, const float* __restrict__ fd,
                              const float* __restrict__ tables,
                              const _Float16* __restrict__ w1, const _Float16* __restrict__ w2,
                              const _Float16* __restrict__ w3, const _Float16* __restrict__ w4,
                              const _Float16* __restrict__ w5,
                              const float* __restrict__ b1, const float* __restrict__ b2,
                              const float* __restrict__ b3, const float* __restrict__ b4,
                              const float* __restrict__ b5,
                              float* __restrict__ out) {
  const int lane = threadIdx.x & 31;
  const int wid  = threadIdx.x >> 5;
  const bool lo  = lane < 16;
  const int ray  = blockIdx.x * 128 + wid * 16 + (lane & 15);

  // warm L2 with first-layer weights while the hash gathers are in flight
  __builtin_prefetch(w1 + lane * 16, 0, 0);
  __builtin_prefetch(w2 + lane * 16, 0, 0);

  // ---- raw inputs (12 feats) -------------------------------------------
  float raw[12];
#pragma unroll
  for (int c = 0; c < 3; ++c) {
    raw[c]     = x [ray * 3 + c];
    raw[3 + c] = wi[ray * 3 + c];
    raw[6 + c] = nn[ray * 3 + c];
    raw[9 + c] = fd[ray * 3 + c];
  }
  const float px = raw[0], py = raw[1], pz = raw[2];

  // ---- hash-grid encode: lane<16 -> levels {0,5,6,7}; lane>=16 -> {1,2,3,4}
  float lv[4][4];
#pragma unroll
  for (int t = 0; t < 4; ++t) {
    const int level = lo ? (t == 0 ? 0 : t + 4) : (t + 1);
    const float res = (float)(16 << level);
    float xs = px * res, ys = py * res, zs = pz * res;
    float fx0 = floorf(xs), fy0 = floorf(ys), fz0 = floorf(zs);
    float fx = xs - fx0, fy = ys - fy0, fz = zs - fz0;
    unsigned cx = (unsigned)fx0, cy = (unsigned)fy0, cz = (unsigned)fz0;
    const float4* tb = reinterpret_cast<const float4*>(tables + (size_t)level * ((size_t)TSIZE * 4));
    float a0 = 0.f, a1 = 0.f, a2 = 0.f, a3 = 0.f;
#pragma unroll
    for (int c = 0; c < 8; ++c) {
      const unsigned ci = (c >> 2) & 1, cj = (c >> 1) & 1, ck = c & 1;
      unsigned h = ((cx + ci) * 1u) ^ ((cy + cj) * P1) ^ ((cz + ck) * P2);
      h &= TMASK;
      float4 f = tb[h];
      float w = (ci ? fx : 1.f - fx) * (cj ? fy : 1.f - fy) * (ck ? fz : 1.f - fz);
      a0 += w * f.x; a1 += w * f.y; a2 += w * f.z; a3 += w * f.w;
    }
    lv[t][0] = a0; lv[t][1] = a1; lv[t][2] = a2; lv[t][3] = a3;
  }

  // ---- layer-1 B fragments (K = 64, 2 frags), feats: 12 raw + 8*4 hash + pad
  v8u B[8], NB[8];
  {
    v16h f0, f1;
#pragma unroll
    for (int h = 0; h < 12; ++h)  // lane<16: raw 0..11 ; lane>=16: L1..L3
      f0[h] = (_Float16)(lo ? raw[h] : lv[h >> 2][h & 3]);
#pragma unroll
    for (int h = 12; h < 16; ++h) // lane<16: L0 ; lane>=16: L4
      f0[h] = (_Float16)(lo ? lv[0][h - 12] : lv[3][h - 12]);
#pragma unroll
    for (int h = 0; h < 16; ++h) { // feats 32..47 (lane<16: L5..L7 + pad) / 48..63 (zero)
      float v = (lo && h < 12) ? lv[1 + (h >> 2)][h & 3] : 0.0f;
      f1[h] = (_Float16)v;
    }
    B[0] = __builtin_bit_cast(v8u, f0);
    B[1] = __builtin_bit_cast(v8u, f1);
  }

  // ---- MLP: 44(->64) -> 256 -> 256 -> 256 -> 256 -> 3 -------------------
  mlp_layer<2>(w1, b1, B, NB, lo, lane);
  mlp_layer<8>(w2, b2, NB, B, lo, lane);
  mlp_layer<8>(w3, b3, B, NB, lo, lane);
  mlp_layer<8>(w4, b4, NB, B, lo, lane);

  // ---- output layer: M tile 0 (rows 0..2 valid), |.|, store --------------
  {
    float o0 = b5[0], o1 = b5[1], o2 = b5[2];
    v8f acc = {lo ? o0 : 0.f, lo ? o1 : 0.f, lo ? o2 : 0.f, 0.f, 0.f, 0.f, 0.f, 0.f};
    v16h wc = ldw(w5, 0, lane);
#pragma unroll
    for (int kt = 0; kt < 8; ++kt) {
      v16h nw = wc;
      if (kt < 7) nw = ldw(w5, kt + 1, lane);
      acc = WMMA_F16(wc, __builtin_bit_cast(v16h, B[kt]), acc);
      wc = nw;
    }
    if (lo) {
      out[ray * 3 + 0] = fabsf(acc[0]);
      out[ray * 3 + 1] = fabsf(acc[1]);
      out[ray * 3 + 2] = fabsf(acc[2]);
    }
  }
}

// ---------------------------------------------------------------------------
extern "C" void kernel_launch(void* const* d_in, const int* in_sizes, int n_in,
                              void* d_out, int out_size, void* d_ws, size_t ws_size,
                              hipStream_t stream) {
  (void)n_in; (void)out_size; (void)ws_size;
  const float* x   = (const float*)d_in[0];
  const float* wi  = (const float*)d_in[1];
  const float* nn  = (const float*)d_in[2];
  const float* fd  = (const float*)d_in[3];
  const float* tbl = (const float*)d_in[4];
  const float* W1 = (const float*)d_in[5],  *b1 = (const float*)d_in[6];
  const float* W2 = (const float*)d_in[7],  *b2 = (const float*)d_in[8];
  const float* W3 = (const float*)d_in[9],  *b3 = (const float*)d_in[10];
  const float* W4 = (const float*)d_in[11], *b4 = (const float*)d_in[12];
  const float* W5 = (const float*)d_in[13], *b5 = (const float*)d_in[14];
  float* out = (float*)d_out;

  const int N = in_sizes[0] / 3;  // 262144; divisible by 128

  _Float16* wsh = (_Float16*)d_ws;
  // fragment-buffer offsets (halves)
  const size_t OFF1 = 0;        // 16 mt * 2 kt * 512 = 16384
  const size_t OFF2 = 16384;    // 16 * 8 * 512 = 65536
  const size_t OFF3 = 81920;
  const size_t OFF4 = 147456;
  const size_t OFF5 = 212992;   // 1 * 8 * 512 = 4096

  swz_weights<<<64,  256, 0, stream>>>(W1, wsh + OFF1, 256, 44,  2, 16384);
  swz_weights<<<256, 256, 0, stream>>>(W2, wsh + OFF2, 256, 256, 8, 65536);
  swz_weights<<<256, 256, 0, stream>>>(W3, wsh + OFF3, 256, 256, 8, 65536);
  swz_weights<<<256, 256, 0, stream>>>(W4, wsh + OFF4, 256, 256, 8, 65536);
  swz_weights<<<16,  256, 0, stream>>>(W5, wsh + OFF5, 3,   256, 8, 4096);

  nrfield_fused<<<N / 128, 256, 0, stream>>>(
      x, wi, nn, fd, tbl,
      wsh + OFF1, wsh + OFF2, wsh + OFF3, wsh + OFF4, wsh + OFF5,
      b1, b2, b3, b4, b5, out);
}
